// SwitchHeadAttention_13013750906911
// MI455X (gfx1250) — matrix operand order repaired and checked by reading, not compiled, not run on hardware
//
#include <hip/hip_runtime.h>
#include <hip/hip_bf16.h>
#include <cstdint>

#define B_   4
#define T_   1024
#define DIM_ 1024
#define H_   8
#define E_   5
#define DH_  64
#define KSEL_ 3
#define BK_  64   // K-slab staged per async step

typedef __bf16 bf16;
typedef __attribute__((ext_vector_type(16))) __bf16 v16bf;
typedef __attribute__((ext_vector_type(8)))  __bf16 v8bf;
typedef __attribute__((ext_vector_type(8)))  float  v8f;

// ---------- helpers ----------

__device__ __forceinline__ bf16 f2bf(float f) {
  unsigned u; __builtin_memcpy(&u, &f, 4);
  u += 0x7FFFu + ((u >> 16) & 1u);          // round-to-nearest-even
  unsigned short hs = (unsigned short)(u >> 16);
  bf16 r; __builtin_memcpy(&r, &hs, 2);
  return r;
}

// A-matrix fragment (16x32 bf16, row-major source, stride ld):
// lane L (hf=L/16, m=L%16): elems 0..7 = K 8*hf..8*hf+7 ; elems 8..15 = K 16+8*hf..+7
__device__ __forceinline__ v16bf load_fragA(const bf16* base, int ld, int lane) {
  int hf = lane >> 4, m = lane & 15;
  const bf16* p = base + (size_t)m * ld + 8 * hf;
  v8bf lo = *(const v8bf*)p;
  v8bf hi = *(const v8bf*)(p + 16);
  v16bf r;
#pragma unroll
  for (int i = 0; i < 8; ++i) { r[i] = lo[i]; r[i + 8] = hi[i]; }
  return r;
}

// B-matrix fragment (32x16 bf16) sourced from TRANSPOSED storage Bt (N x K row-major):
// lane L holds column n=L%16: elems 0..7 = K 8*hf..8*hf+7 ; elems 8..15 = K 16+8*hf..+7 (Bt row n).
__device__ __forceinline__ v16bf load_fragB(const bf16* base, int ld, int lane) {
  int hf = lane >> 4, n = lane & 15;
  const bf16* p = base + (size_t)n * ld + 8 * hf;
  v8bf lo = *(const v8bf*)p;
  v8bf hi = *(const v8bf*)(p + 16);
  v16bf r;
#pragma unroll
  for (int i = 0; i < 8; ++i) { r[i] = lo[i]; r[i + 8] = hi[i]; }
  return r;
}

__device__ __forceinline__ v8f wmma_bf16(v16bf a, v16bf b, v8f c) {
  return __builtin_amdgcn_wmma_f32_16x16x32_bf16(
      /*neg_a=*/false, a, /*neg_b=*/false, b,
      /*c_mod=*/(short)0, c, /*reuse_a=*/false, /*reuse_b=*/false);
}

// CDNA5 async global->LDS copy (ASYNCcnt-tracked), 16 bytes per lane.
__device__ __forceinline__ void async_g2l_b128(unsigned lds_byte_addr, const bf16* gaddr) {
  asm volatile("global_load_async_to_lds_b128 %0, %1, off"
               :: "v"(lds_byte_addr), "v"(gaddr) : "memory");
}

// ---------- conversion / packing kernels ----------

__global__ void cvt_bf16_kernel(const float* __restrict__ src, bf16* __restrict__ dst, int n) {
  int idx = blockIdx.x * blockDim.x + threadIdx.x;
  if (idx < n) dst[idx] = f2bf(src[idx]);
}

// src (Kd x N) row-major f32 -> dst (N x Kd) row-major bf16
__global__ void transpose_cvt_kernel(const float* __restrict__ src, bf16* __restrict__ dst,
                                     int Kd, int N) {
  int idx = blockIdx.x * blockDim.x + threadIdx.x;
  if (idx >= N * Kd) return;
  int k = idx % Kd, nn = idx / Kd;
  dst[idx] = f2bf(src[(size_t)k * N + nn]);
}

// Build padded transposed [Ws | Wd | 0] -> dst (128 x 1024) bf16
__global__ void build_wsd_kernel(const float* __restrict__ Ws, const float* __restrict__ Wd,
                                 bf16* __restrict__ dst) {
  int idx = blockIdx.x * blockDim.x + threadIdx.x;
  if (idx >= 128 * 1024) return;
  int k = idx % 1024, nn = idx / 1024;
  float v = 0.f;
  if (nn < 40)      v = Ws[(size_t)k * 40 + nn];
  else if (nn < 80) v = Wd[(size_t)k * 40 + (nn - 40)];
  dst[idx] = f2bf(v);
}

// Wo (64 x 5120) -> Wo_t (1024 x 320) bf16 : Wo_t[m][e*64+d] = Wo[d][e*1024+m]
__global__ void build_wo_kernel(const float* __restrict__ Wo, bf16* __restrict__ dst) {
  int idx = blockIdx.x * blockDim.x + threadIdx.x;
  if (idx >= 1024 * 320) return;
  int j = idx % 320, m = idx / 320;
  int e = j / 64, d = j % 64;
  dst[idx] = f2bf(Wo[(size_t)d * 5120 + (size_t)e * 1024 + m]);
}

// ---------- generic bf16 WMMA GEMM : C(MxN) f32 = A(MxK) bf16 * Bt(NxK) bf16 ----------
// 8 waves/block; block tile 64x128; wave tile 16x64 (4 C tiles).
// A block-tile staged to LDS with double-buffered async DMA (ASYNCcnt path);
// B fragments stream from global (weights are L2-resident on the 192MB L2).
__global__ __launch_bounds__(256) void gemm_bf16_kernel(
    const bf16* __restrict__ A, const bf16* __restrict__ Bt, float* __restrict__ C,
    int M, int N, int Kd) {
  __shared__ __align__(16) bf16 Atile[2][64 * BK_];   // 2 x 8KB
  int tid  = threadIdx.x;
  int lane = tid & 31;
  int wave = tid >> 5;
  int wrow = wave >> 1, wcol = wave & 1;
  int row0 = blockIdx.y * 64;
  int col0 = blockIdx.x * 128 + wcol * 64;

  // async stage mapping: 256 threads x 2 x 16B = 8KB = 64 rows x 64 cols bf16
  int ar = tid >> 3;           // 0..31 (and ar+32)
  int ac = (tid & 7) * 8;      // col element offset, 16B granules
  const bf16* ga = A + (size_t)(row0 + ar) * Kd + ac;
  const size_t arow32 = (size_t)32 * Kd;

  v8f acc[4] = {};

  // prologue: stage slab 0 into buffer 0
  async_g2l_b128((unsigned)(uintptr_t)&Atile[0][ar * BK_ + ac],        ga);
  async_g2l_b128((unsigned)(uintptr_t)&Atile[0][(ar + 32) * BK_ + ac], ga + arow32);

  for (int kc = 0; kc < Kd; kc += BK_) {
    int buf = (kc / BK_) & 1;
    bool has_next = (kc + BK_) < Kd;
    if (has_next) {   // prefetch next slab into the other buffer
      async_g2l_b128((unsigned)(uintptr_t)&Atile[buf ^ 1][ar * BK_ + ac],
                     ga + (kc + BK_));
      async_g2l_b128((unsigned)(uintptr_t)&Atile[buf ^ 1][(ar + 32) * BK_ + ac],
                     ga + (kc + BK_) + arow32);
      asm volatile("s_wait_asynccnt 0x2" ::: "memory");   // current slab landed
    } else {
      asm volatile("s_wait_asynccnt 0x0" ::: "memory");
    }
    __syncthreads();

    const bf16* at = &Atile[buf][(size_t)wrow * 16 * BK_];
    v16bf a0 = load_fragA(at,      BK_, lane);
    v16bf a1 = load_fragA(at + 32, BK_, lane);
#pragma unroll
    for (int tt = 0; tt < 4; ++tt) {
      const bf16* bb = Bt + (size_t)(col0 + tt * 16) * Kd + kc;
      v16bf b0 = load_fragB(bb,      Kd, lane);
      v16bf b1 = load_fragB(bb + 32, Kd, lane);
      acc[tt] = wmma_bf16(a0, b0, acc[tt]);
      acc[tt] = wmma_bf16(a1, b1, acc[tt]);
    }
    __syncthreads();   // all waves done with buf before it is re-staged
  }

  int hf = lane >> 4, n = lane & 15;
#pragma unroll
  for (int tt = 0; tt < 4; ++tt)
#pragma unroll
    for (int r = 0; r < 8; ++r)
      C[(size_t)(row0 + wrow * 16 + r + 8 * hf) * N + col0 + tt * 16 + n] = acc[tt][r];
}

// ---------- gating: sigmoid + top-3-of-5 masks ----------
__global__ void gate_kernel(const float* __restrict__ ssd,
                            float* __restrict__ ssg, float* __restrict__ sdg,
                            float* __restrict__ mdm) {
  int idx = blockIdx.x * blockDim.x + threadIdx.x;   // over B*T*H
  if (idx >= B_ * T_ * H_) return;
  int h = idx % H_, bt = idx / H_;
  const float* row = ssd + (size_t)bt * 128;
  float ss[E_], sd[E_];
#pragma unroll
  for (int e = 0; e < E_; ++e) {
    ss[e] = 1.f / (1.f + __expf(-row[h * E_ + e]));
    sd[e] = 1.f / (1.f + __expf(-row[40 + h * E_ + e]));
  }
#pragma unroll
  for (int e = 0; e < E_; ++e) {
    int cs = 0, cd = 0;
#pragma unroll
    for (int j = 0; j < E_; ++j) {
      if (ss[j] > ss[e] || (ss[j] == ss[e] && j < e)) cs++;
      if (sd[j] > sd[e] || (sd[j] == sd[e] && j < e)) cd++;
    }
    float ms = (cs < KSEL_) ? 1.f : 0.f;
    float md = (cd < KSEL_) ? 1.f : 0.f;
    size_t o = (size_t)(bt * H_ + h) * E_ + e;
    ssg[o] = ss[e] * ms;
    sdg[o] = sd[e] * md;
    mdm[o] = md;
  }
}

// ---------- expert-weighted combine -> q,k (B,H,T,DH) bf16 ; v transposed (B,H,DH,T) ----------
__global__ void combine_qkv_kernel(const float* __restrict__ q5, const float* __restrict__ kv5,
                                   const float* __restrict__ sdg, const float* __restrict__ ssg,
                                   bf16* __restrict__ qT, bf16* __restrict__ kT,
                                   bf16* __restrict__ vT) {
  int idx = blockIdx.x * blockDim.x + threadIdx.x;   // B*H*T*DH
  if (idx >= B_ * H_ * T_ * DH_) return;
  int d = idx & 63;
  int rest = idx >> 6;
  int t = rest & (T_ - 1);
  int rest2 = rest >> 10;
  int h = rest2 & (H_ - 1);
  int b = rest2 >> 3;
  int bt = b * T_ + t;
  const float* gsd = sdg + (size_t)(bt * H_ + h) * E_;
  const float* gss = ssg + (size_t)(bt * H_ + h) * E_;
  float q = 0.f, k = 0.f, v = 0.f;
#pragma unroll
  for (int e = 0; e < E_; ++e) {
    q += gsd[e] * q5[(size_t)bt * 2560 + (size_t)(h * E_ + e) * 64 + d];
    k += gss[e] * kv5[(size_t)bt * 5120 + (size_t)(h * E_ + e) * 64 + d];
    v += gss[e] * kv5[(size_t)bt * 5120 + (size_t)((H_ + h) * E_ + e) * 64 + d];
  }
  size_t oq = ((size_t)(b * H_ + h) * T_ + t) * DH_ + d;
  qT[oq] = f2bf(q);
  kT[oq] = f2bf(k);
  vT[((size_t)(b * H_ + h) * DH_ + d) * T_ + t] = f2bf(v);
}

// ---------- flash attention: 1 wave / 16 query rows of one (b,h) ----------
__global__ __launch_bounds__(32) void attn_kernel(const bf16* __restrict__ qT,
                                                  const bf16* __restrict__ kT,
                                                  const bf16* __restrict__ vT,
                                                  float* __restrict__ outb) {
  __shared__ __align__(16) bf16 Pst[16 * 32];   // P tile staging: C-layout -> A-layout
  int lane = threadIdx.x;
  int hf = lane >> 4, l16 = lane & 15;
  int i0 = blockIdx.x * 16;
  int h = blockIdx.y, bb = blockIdx.z;
  const bf16* qbase = qT + (((size_t)bb * H_ + h) * T_ + i0) * DH_;
  const bf16* kbase = kT + (((size_t)bb * H_ + h) * T_) * DH_;
  const bf16* vbase = vT + (((size_t)bb * H_ + h) * DH_) * T_;
  const float scale = 0.125f;   // DH^-0.5

  v16bf aq0 = load_fragA(qbase, DH_, lane);        // d = 0..31
  v16bf aq1 = load_fragA(qbase + 32, DH_, lane);   // d = 32..63

  v8f acc[4] = {};
  float mrow[8], lrow[8];
#pragma unroll
  for (int r = 0; r < 8; ++r) { mrow[r] = -3.0e38f; lrow[r] = 0.f; }

  for (int j0 = 0; j0 < T_; j0 += 32) {
    // --- S tiles (16x16 each) via WMMA ---
    v8f c0 = {}, c1 = {};
    {
      v16bf bk = load_fragB(kbase + (size_t)j0 * DH_, DH_, lane);
      c0 = wmma_bf16(aq0, bk, c0);
      bk = load_fragB(kbase + (size_t)j0 * DH_ + 32, DH_, lane);
      c0 = wmma_bf16(aq1, bk, c0);
      bk = load_fragB(kbase + (size_t)(j0 + 16) * DH_, DH_, lane);
      c1 = wmma_bf16(aq0, bk, c1);
      bk = load_fragB(kbase + (size_t)(j0 + 16) * DH_ + 32, DH_, lane);
      c1 = wmma_bf16(aq1, bk, c1);
    }
    // --- online softmax row statistics (rows r+8*hf live in the lane's half) ---
    float e0[8], e1[8], corr[8];
#pragma unroll
    for (int r = 0; r < 8; ++r) {
      float s0 = c0[r] * scale, s1 = c1[r] * scale;
      float tmx = fmaxf(s0, s1);
#pragma unroll
      for (int off = 8; off >= 1; off >>= 1)
        tmx = fmaxf(tmx, __shfl_xor(tmx, off, 32));
      float mn = fmaxf(mrow[r], tmx);
      float cr = __expf(mrow[r] - mn);
      e0[r] = __expf(s0 - mn);
      e1[r] = __expf(s1 - mn);
      float sm = e0[r] + e1[r];
#pragma unroll
      for (int off = 8; off >= 1; off >>= 1)
        sm += __shfl_xor(sm, off, 32);
      lrow[r] = lrow[r] * cr + sm;
      mrow[r] = mn;
      corr[r] = cr;
    }
    // --- stage P tile to LDS (convert C layout -> A layout) ---
    __syncthreads();
#pragma unroll
    for (int r = 0; r < 8; ++r) {
      Pst[(r + 8 * hf) * 32 + l16]      = f2bf(e0[r]);
      Pst[(r + 8 * hf) * 32 + 16 + l16] = f2bf(e1[r]);
    }
    __syncthreads();
    // --- rescale accumulators, then O += P @ V ---
#pragma unroll
    for (int tt = 0; tt < 4; ++tt)
#pragma unroll
      for (int r = 0; r < 8; ++r) acc[tt][r] *= corr[r];
    v16bf ap = load_fragA(Pst, 32, lane);
#pragma unroll
    for (int tt = 0; tt < 4; ++tt) {
      v16bf bv = load_fragB(vbase + (size_t)(tt * 16) * T_ + j0, T_, lane);
      acc[tt] = wmma_bf16(ap, bv, acc[tt]);
    }
  }
  // --- epilogue: divide by row sums, write out (B,T,H,DH) f32 ---
#pragma unroll
  for (int tt = 0; tt < 4; ++tt)
#pragma unroll
    for (int r = 0; r < 8; ++r) {
      int row = i0 + r + 8 * hf;
      outb[(((size_t)bb * T_ + row) * H_ + h) * DH_ + tt * 16 + l16] =
          acc[tt][r] / lrow[r];
    }
}

// ---------- fold md gate: u[bt, e*64+d] = sum_h md * out[bt,h,d] (bf16) ----------
__global__ void u_kernel(const float* __restrict__ outb, const float* __restrict__ mdm,
                         bf16* __restrict__ u) {
  int idx = blockIdx.x * blockDim.x + threadIdx.x;   // B*T*E*DH
  if (idx >= B_ * T_ * E_ * DH_) return;
  int d = idx & 63;
  int rest = idx >> 6;
  int e = rest % E_;
  int bt = rest / E_;
  float s = 0.f;
#pragma unroll
  for (int h = 0; h < H_; ++h)
    s += mdm[(size_t)(bt * H_ + h) * E_ + e] * outb[(size_t)(bt * H_ + h) * DH_ + d];
  u[(size_t)bt * 320 + e * 64 + d] = f2bf(s);
}

// ---------- launcher ----------
extern "C" void kernel_launch(void* const* d_in, const int* in_sizes, int n_in,
                              void* d_out, int out_size, void* d_ws, size_t ws_size,
                              hipStream_t stream) {
  const float* x   = (const float*)d_in[0];
  const float* Wq  = (const float*)d_in[1];
  const float* Wkv = (const float*)d_in[2];
  const float* Ws  = (const float*)d_in[3];
  const float* Wd  = (const float*)d_in[4];
  const float* Wo  = (const float*)d_in[5];
  float* out = (float*)d_out;
  (void)in_sizes; (void)n_in; (void)out_size; (void)ws_size;

  size_t off = 0;
  auto alloc = [&](size_t bytes) {
    void* p = (char*)d_ws + off;
    off += (bytes + 255) & ~(size_t)255;
    return p;
  };
  const int MT = B_ * T_;   // 4096
  bf16*  xb    = (bf16*) alloc((size_t)MT * DIM_ * 2);
  bf16*  Wq_t  = (bf16*) alloc((size_t)2560 * 1024 * 2);
  bf16*  Wkv_t = (bf16*) alloc((size_t)5120 * 1024 * 2);
  bf16*  Wsd_t = (bf16*) alloc((size_t)128 * 1024 * 2);
  bf16*  Wo_t  = (bf16*) alloc((size_t)1024 * 320 * 2);
  float* ssd   = (float*)alloc((size_t)MT * 128 * 4);
  float* ssg   = (float*)alloc((size_t)MT * H_ * E_ * 4);
  float* sdg   = (float*)alloc((size_t)MT * H_ * E_ * 4);
  float* mdm   = (float*)alloc((size_t)MT * H_ * E_ * 4);
  float* q5    = (float*)alloc((size_t)MT * 2560 * 4);
  float* kv5   = (float*)alloc((size_t)MT * 5120 * 4);
  bf16*  qT    = (bf16*) alloc((size_t)B_ * H_ * T_ * DH_ * 2);
  bf16*  kT    = (bf16*) alloc((size_t)B_ * H_ * T_ * DH_ * 2);
  bf16*  vT    = (bf16*) alloc((size_t)B_ * H_ * T_ * DH_ * 2);
  float* outb  = (float*)alloc((size_t)MT * H_ * DH_ * 4);
  bf16*  ub    = (bf16*) alloc((size_t)MT * 320 * 2);

  auto g1 = [](int n) { return dim3((n + 255) / 256); };

  // 1. bf16 conversions / weight packing
  { int n = MT * DIM_;       cvt_bf16_kernel<<<g1(n), 256, 0, stream>>>(x, xb, n); }
  { int n = 2560 * 1024;     transpose_cvt_kernel<<<g1(n), 256, 0, stream>>>(Wq, Wq_t, 1024, 2560); }
  { int n = 5120 * 1024;     transpose_cvt_kernel<<<g1(n), 256, 0, stream>>>(Wkv, Wkv_t, 1024, 5120); }
  { int n = 128 * 1024;      build_wsd_kernel<<<g1(n), 256, 0, stream>>>(Ws, Wd, Wsd_t); }
  { int n = 1024 * 320;      build_wo_kernel<<<g1(n), 256, 0, stream>>>(Wo, Wo_t); }

  // 2. gating scores GEMM (4096 x 128 x 1024), then gate masks
  gemm_bf16_kernel<<<dim3(128 / 128, MT / 64), 256, 0, stream>>>(xb, Wsd_t, ssd, MT, 128, 1024);
  { int n = MT * H_;         gate_kernel<<<g1(n), 256, 0, stream>>>(ssd, ssg, sdg, mdm); }

  // 3. projection GEMMs
  gemm_bf16_kernel<<<dim3(2560 / 128, MT / 64), 256, 0, stream>>>(xb, Wq_t, q5, MT, 2560, 1024);
  gemm_bf16_kernel<<<dim3(5120 / 128, MT / 64), 256, 0, stream>>>(xb, Wkv_t, kv5, MT, 5120, 1024);

  // 4. expert-weighted combine -> q,k,(v transposed)
  { int n = B_ * H_ * T_ * DH_;
    combine_qkv_kernel<<<g1(n), 256, 0, stream>>>(q5, kv5, sdg, ssg, qT, kT, vT); }

  // 5. flash attention
  attn_kernel<<<dim3(T_ / 16, H_, B_), 32, 0, stream>>>(qT, kT, vT, outb);

  // 6. fold md gate, final output GEMM (4096 x 1024 x 320) -> d_out
  { int n = B_ * T_ * E_ * DH_; u_kernel<<<g1(n), 256, 0, stream>>>(outb, mdm, ub); }
  gemm_bf16_kernel<<<dim3(DIM_ / 128, MT / 64), 256, 0, stream>>>(ub, Wo_t, out, MT, DIM_, 320);
}
